// Block_30674656428363
// MI455X (gfx1250) — matrix operand order repaired and checked by reading, not compiled, not run on hardware
//
#include <hip/hip_runtime.h>
#include <hip/hip_bf16.h>

// ---------------------------------------------------------------------------
// Transformer block for MI455X (gfx1250).
//  * bf16 WMMA 16x16x32, fp32 accumulation (v_wmma_f32_16x16x32_bf16)
//  * K-outer GEMM loops: 4 independent WMMA accumulator chains per wave
//  * weight/x staging via GLOBAL_LOAD_ASYNC_TO_LDS_B128 (ASYNCcnt) when the
//    builtin exists, with double-buffered panel prefetch; sync-copy fallback
//  * LN / softmax / router / residual stream in fp32
// One 256-thread workgroup (8 wave32) per batch element.
// ---------------------------------------------------------------------------

typedef __attribute__((ext_vector_type(16))) __bf16 v16bf;
typedef __attribute__((ext_vector_type(8)))  __bf16 v8bf;
typedef __attribute__((ext_vector_type(8)))  float  v8f;
typedef __attribute__((ext_vector_type(4)))  int    v4i;

#define TT   64
#define CC   128
#define HH   4
#define HS   32
#define LDA  136   // bf16 row stride for 128-col buffers (16B aligned, bank-skewed)
#define LDX  132   // f32 row stride for residual buffer
#define LDSC 68    // f32 row stride for scores
#define LDP  72    // bf16 row stride for probs / vT
#define ATT_SCALE 0.08838834764831845f   // 1/sqrt(C), C = 128 (as in reference)

// ws (bf16) element offsets: all weights pre-transposed to B^T = [N][K] row-major
#define WQT   0
#define WKT   16384
#define WVT   32768
#define WPRJT 49152
#define W1T   65536     // [512][128]
#define W2T   131072    // [128][512]
#define WS_ELEMS 196608

// ---------------- async global->LDS (CDNA5) with graceful fallback ---------
#if defined(__has_builtin)
#if __has_builtin(__builtin_amdgcn_global_load_async_to_lds_b128) && \
    __has_builtin(__builtin_amdgcn_s_wait_asynccnt)
#define HAVE_ASYNC 1
#endif
#endif
#ifndef HAVE_ASYNC
#define HAVE_ASYNC 0
#endif

__device__ __forceinline__ void async_copy_b128(const void* gsrc, void* ldst) {
#if HAVE_ASYNC
    typedef __attribute__((address_space(1))) v4i v4i_g;   // global (prints as __device__)
    typedef __attribute__((address_space(3))) v4i v4i_l;   // LDS
    // Address-space-typed pointers built via integer casts: AS1 = 64-bit flat
    // address; AS3 = low 32 bits of the flat LDS address (aperture in MSBs).
    v4i_g* g = (v4i_g*)(unsigned long long)(size_t)gsrc;
    v4i_l* l = (v4i_l*)(unsigned int)(size_t)ldst;
    __builtin_amdgcn_global_load_async_to_lds_b128(g, l, 0, 0);
#else
    *(int4*)ldst = *(const int4*)gsrc;
#endif
}

__device__ __forceinline__ void async_wait_all() {
#if HAVE_ASYNC
    __builtin_amdgcn_s_wait_asynccnt(0);
#endif
}

// copy 128 x 128 bf16 panel, global(row stride src_ld) -> LDS(row stride LDA)
__device__ __forceinline__ void stage_panel(__bf16* dst, const __bf16* src, int src_ld) {
    for (int v = threadIdx.x; v < 128 * 128 / 8; v += 256) {
        int e = v * 8;
        int r = e >> 7, c0 = e & 127;
        async_copy_b128(&src[r * src_ld + c0], &dst[r * LDA + c0]);
    }
}

// ---------------- WMMA fragment helpers ------------------------------------
union F16x16 { v16bf v; v8bf h[2]; };

// A fragment 16x32 bf16 from row-major [rows][ld]:
// lane L (r=L&15, hi=L>>4): row r0+r, K = {hi*8..+7, 16+hi*8..+7}
__device__ __forceinline__ v16bf load_frag_a(const __bf16* buf, int ld, int r0, int k0) {
    int lane = threadIdx.x & 31;
    const __bf16* p = buf + (r0 + (lane & 15)) * ld + k0 + ((lane >> 4) << 3);
    F16x16 f;
    f.h[0] = *(const v8bf*)p;
    f.h[1] = *(const v8bf*)(p + 16);
    return f.v;
}

// B fragment 32x16 bf16 from B^T row-major [N][ld]:
// lane L: column n0+(L&15), K = {hi*8..+7, 16+hi*8..+7}
__device__ __forceinline__ v16bf load_frag_b(const __bf16* bufT, int ld, int n0, int k0) {
    int lane = threadIdx.x & 31;
    const __bf16* p = bufT + (n0 + (lane & 15)) * ld + k0 + ((lane >> 4) << 3);
    F16x16 f;
    f.h[0] = *(const v8bf*)p;
    f.h[1] = *(const v8bf*)(p + 16);
    return f.v;
}

__device__ __forceinline__ v8f wmma_bf16(v16bf a, v16bf b, v8f c) {
    return __builtin_amdgcn_wmma_f32_16x16x32_bf16(false, a, false, b, (short)0, c, false, false);
}

// D element r -> (M = m0 + (lane>>4)*8 + r, N = n0 + (lane&15))
__device__ __forceinline__ void store_frag_bf16(__bf16* buf, int ld, int m0, int n0, v8f d) {
    int lane = threadIdx.x & 31;
    int m = m0 + ((lane >> 4) << 3);
    int n = n0 + (lane & 15);
#pragma unroll
    for (int r = 0; r < 8; ++r) buf[(m + r) * ld + n] = (__bf16)d[r];
}

// transposed store: element (M,N) -> bufT[N][M] (contiguous in M -> b128)
__device__ __forceinline__ void store_frag_bf16_T(__bf16* bufT, int ldT, int m0, int n0, v8f d) {
    int lane = threadIdx.x & 31;
    int m = m0 + ((lane >> 4) << 3);
    __bf16* p = &bufT[(n0 + (lane & 15)) * ldT + m];
#pragma unroll
    for (int r = 0; r < 8; ++r) p[r] = (__bf16)d[r];
}

// [64 x 128] = A[64 x 128] * W^T ; wave `wv` computes column block wv*16..+15,
// all 4 row tiles. K-outer: one B frag shared by 4 independent WMMA chains.
__device__ __forceinline__ void gemm_64x128(const __bf16* A, const __bf16* W,
                                            v8f acc[4], int wv) {
#pragma unroll
    for (int kk = 0; kk < 4; ++kk) {
        v16bf bb = load_frag_b(W, LDA, wv * 16, kk * 32);
#pragma unroll
        for (int tm = 0; tm < 4; ++tm) {
            v16bf a = load_frag_a(A, LDA, tm * 16, kk * 32);
            acc[tm] = wmma_bf16(a, bb, acc[tm]);
        }
    }
}

#define ZERO4(acc) do { \
    _Pragma("unroll") \
    for (int _i = 0; _i < 4; ++_i) \
        acc[_i] = (v8f){0.f,0.f,0.f,0.f,0.f,0.f,0.f,0.f}; \
} while (0)

// ---------------------------------------------------------------------------
// prep: convert + transpose all weights to bf16 B^T layout in d_ws
// ---------------------------------------------------------------------------
__global__ void prep_weights(const float* __restrict__ wq, const float* __restrict__ wk,
                             const float* __restrict__ wv, const float* __restrict__ wproj,
                             const float* __restrict__ w1, const float* __restrict__ w2,
                             __bf16* __restrict__ ws) {
    int i = blockIdx.x * blockDim.x + threadIdx.x;
    if (i >= WS_ELEMS) return;
    float val;
    if (i < WKT) {                       // wqT[n=h*32+d][c] = wq[h][c][d]
        int n = i >> 7, c = i & 127, h = n >> 5, d = n & 31;
        val = wq[(h * 128 + c) * 32 + d];
    } else if (i < WVT) {
        int j = i - WKT; int n = j >> 7, c = j & 127, h = n >> 5, d = n & 31;
        val = wk[(h * 128 + c) * 32 + d];
    } else if (i < WPRJT) {
        int j = i - WVT; int n = j >> 7, c = j & 127, h = n >> 5, d = n & 31;
        val = wv[(h * 128 + c) * 32 + d];
    } else if (i < W1T) {                // wprojT[n][c] = w_proj[c][n]
        int j = i - WPRJT; int n = j >> 7, c = j & 127;
        val = wproj[c * 128 + n];
    } else if (i < W2T) {                // w1T[n][k] = w1[k][n], [512][128]
        int j = i - W1T; int n = j >> 7, k = j & 127;
        val = w1[k * 512 + n];
    } else {                             // w2T[n][k] = w2[k][n], [128][512]
        int j = i - W2T; int n = j >> 9, k = j & 511;
        val = w2[k * 128 + n];
    }
    ws[i] = (__bf16)val;
}

// ---------------------------------------------------------------------------
// main block kernel: one workgroup per batch element
// ---------------------------------------------------------------------------
__global__ __launch_bounds__(256) void block_kernel(
        const float* __restrict__ x,
        const float* __restrict__ ln1_g, const float* __restrict__ ln1_b,
        const float* __restrict__ b_proj,
        const float* __restrict__ ln2_g, const float* __restrict__ ln2_b,
        const float* __restrict__ b1,    const float* __restrict__ b2,
        const float* __restrict__ rw1,   const float* __restrict__ rb1,
        const float* __restrict__ rw2,   const float* __restrict__ rb2,
        const float* __restrict__ eb,    const float* __restrict__ bscale,
        const __bf16* __restrict__ ws,   float* __restrict__ out) {

    // ---- manual LDS layout (aliasing is phase-disjoint) -------------------
    // [      0,  33792) xf    f32 [64][132]  residual stream
    // [  33792,  51200) abuf  bf16 [64][136] xn / att-concat / LN2 out
    // [  51200,  68608) qbuf  bf16 [64][136] q   (FFN: hbuf)
    // [  68608,  86016) kbuf  bf16 [64][136] k
    // [  86016, 104448) vT    bf16 [128][72] v^T
    // [ 104448, 139264) wp0   weight panel   (attn: sbuf f32 + pbuf bf16)
    // [ 139264, 174080) wp1   weight panel
    // [ 174080, 208896) wp2   weight panel
    // [ 208896, 243712) wp3   weight panel
    // [ 243712, 245056) rscr  f32 [336] router scratch
    __shared__ __align__(16) unsigned char smem[245056];
    float*  xf   = (float*)smem;
    __bf16* abuf = (__bf16*)(smem + 33792);
    __bf16* qbuf = (__bf16*)(smem + 51200);
    __bf16* kbuf = (__bf16*)(smem + 68608);
    __bf16* vT   = (__bf16*)(smem + 86016);
    __bf16* wp0  = (__bf16*)(smem + 104448);
    __bf16* wp1  = (__bf16*)(smem + 139264);
    __bf16* wp2  = (__bf16*)(smem + 174080);
    __bf16* wp3  = (__bf16*)(smem + 208896);
    __bf16* wpan[4] = {wp0, wp1, wp2, wp3};
    float*  sbuf = (float*)(smem + 104448);     // attn only (64x68 f32)
    __bf16* pbuf = (__bf16*)(smem + 121856);    // attn only (64x72 bf16)
    float*  rscr = (float*)(smem + 243712);

    const int tid  = threadIdx.x;
    const int lane = tid & 31;
    const int wv   = tid >> 5;          // wave id 0..7
    const int bidx = blockIdx.x;
    const float* xb = x + (size_t)bidx * TT * CC;

    // ---- phase 0: async-stage x tile + Q/K weights ------------------------
    for (int v = tid; v < TT * CC / 4; v += 256) {     // f32, 4 per 16B
        int e = v * 4, r = e >> 7, c = e & 127;
        async_copy_b128(&xb[r * 128 + c], &xf[r * LDX + c]);
    }
    stage_panel(wp0, ws + WQT, 128);
    stage_panel(wp1, ws + WKT, 128);
    async_wait_all();
    __syncthreads();

    // ---- LN1 -> abuf (4 threads per row) ---------------------------------
    {
        int row = tid >> 2, part = tid & 3;
        float s = 0.f, s2 = 0.f;
        for (int j = part * 32; j < part * 32 + 32; ++j) {
            float v = xf[row * LDX + j]; s += v; s2 += v * v;
        }
        s  += __shfl_xor(s, 1);  s  += __shfl_xor(s, 2);
        s2 += __shfl_xor(s2, 1); s2 += __shfl_xor(s2, 2);
        float m = s * (1.f / CC), var = s2 * (1.f / CC) - m * m;
        float rstd = rsqrtf(var + 1e-5f);
        for (int j = part * 32; j < part * 32 + 32; ++j)
            abuf[row * LDA + j] = (__bf16)((xf[row * LDX + j] - m) * rstd * ln1_g[j] + ln1_b[j]);
    }
    stage_panel(wp2, ws + WVT, 128);    // V weights land during Q/K compute
    __syncthreads();

    // ---- phase 1: Q and K -------------------------------------------------
    {
        v8f acc[4]; ZERO4(acc);
        gemm_64x128(abuf, wp0, acc, wv);
#pragma unroll
        for (int tm = 0; tm < 4; ++tm) store_frag_bf16(qbuf, LDA, tm * 16, wv * 16, acc[tm]);
    }
    {
        v8f acc[4]; ZERO4(acc);
        gemm_64x128(abuf, wp1, acc, wv);
#pragma unroll
        for (int tm = 0; tm < 4; ++tm) store_frag_bf16(kbuf, LDA, tm * 16, wv * 16, acc[tm]);
    }
    async_wait_all();
    __syncthreads();

    // ---- V (stored transposed); proj weights prefetch during attention ----
    {
        v8f acc[4]; ZERO4(acc);
        gemm_64x128(abuf, wp2, acc, wv);
#pragma unroll
        for (int tm = 0; tm < 4; ++tm) store_frag_bf16_T(vT, LDP, tm * 16, wv * 16, acc[tm]);
    }
    stage_panel(wp3, ws + WPRJT, 128);
    __syncthreads();                    // vT visible; wp0 becomes sbuf/pbuf

    // ---- phase 2: attention per head -------------------------------------
    for (int h = 0; h < HH; ++h) {
        // scores = q @ k^T : 2 tiles/wave sharing the A fragment
        {
            int tile0 = wv * 2;
            int tm = tile0 >> 2, ts0 = tile0 & 3, ts1 = ts0 + 1;
            v16bf a  = load_frag_a(qbuf, LDA, tm * 16, h * HS);
            v16bf b0 = load_frag_b(kbuf, LDA, ts0 * 16, h * HS);
            v16bf b1f = load_frag_b(kbuf, LDA, ts1 * 16, h * HS);
            v8f z = {0.f,0.f,0.f,0.f,0.f,0.f,0.f,0.f};
            v8f d0 = wmma_bf16(a, b0, z);
            v8f d1 = wmma_bf16(a, b1f, z);
            int m = tm * 16 + ((lane >> 4) << 3);
            int c0 = ts0 * 16 + (lane & 15), c1 = ts1 * 16 + (lane & 15);
#pragma unroll
            for (int r = 0; r < 8; ++r) {
                sbuf[(m + r) * LDSC + c0] = d0[r];
                sbuf[(m + r) * LDSC + c1] = d1[r];
            }
        }
        __syncthreads();
        // masked softmax (row-wise, 4 threads per row)
        {
            int row = tid >> 2, part = tid & 3;
            float vals[16];
            float mx = -3.0e38f;
#pragma unroll
            for (int j = 0; j < 16; ++j) {
                int s_ = part * 16 + j;
                float vv = sbuf[row * LDSC + s_] * ATT_SCALE;
                if (s_ > row) vv = -3.0e38f;
                vals[j] = vv;
                mx = fmaxf(mx, vv);
            }
            mx = fmaxf(mx, __shfl_xor(mx, 1));
            mx = fmaxf(mx, __shfl_xor(mx, 2));
            float sum = 0.f;
#pragma unroll
            for (int j = 0; j < 16; ++j) { float e = __expf(vals[j] - mx); vals[j] = e; sum += e; }
            sum += __shfl_xor(sum, 1);
            sum += __shfl_xor(sum, 2);
            float inv = 1.f / sum;
#pragma unroll
            for (int j = 0; j < 16; ++j)
                pbuf[row * LDP + part * 16 + j] = (__bf16)(vals[j] * inv);
        }
        __syncthreads();
        // att = P @ v  (K = 64 -> 2 WMMAs; 8 tiles, one per wave)
        {
            int tm = wv >> 1, tn = wv & 1;
            v8f acc = {0.f,0.f,0.f,0.f,0.f,0.f,0.f,0.f};
#pragma unroll
            for (int kk = 0; kk < 2; ++kk) {
                v16bf a  = load_frag_a(pbuf, LDP, tm * 16, kk * 32);
                v16bf bb = load_frag_b(vT,   LDP, h * HS + tn * 16, kk * 32);
                acc = wmma_bf16(a, bb, acc);
            }
            store_frag_bf16(abuf, LDA, tm * 16, h * HS + tn * 16, acc);
        }
        __syncthreads();
    }
    async_wait_all();                   // wproj (wp3) in place
    __syncthreads();

    // ---- phase 3: proj + residual; FFN panel 0 prefetch -------------------
    stage_panel(wp0, ws + W1T, 128);            // w1 panel 0 (wp0 now free)
    stage_panel(wp1, ws + W2T, 512);            // w2 panel 0
    {
        v8f acc[4]; ZERO4(acc);
        gemm_64x128(abuf, wp3, acc, wv);
        int n = wv * 16 + (lane & 15);
        float bp = b_proj[n];
#pragma unroll
        for (int tm = 0; tm < 4; ++tm) {
            int m = tm * 16 + ((lane >> 4) << 3);
#pragma unroll
            for (int r = 0; r < 8; ++r) xf[(m + r) * LDX + n] += acc[tm][r] + bp;
        }
    }
    async_wait_all();
    __syncthreads();

    // ---- phase 4: LN2 -> abuf --------------------------------------------
    {
        int row = tid >> 2, part = tid & 3;
        float s = 0.f, s2 = 0.f;
        for (int j = part * 32; j < part * 32 + 32; ++j) {
            float v = xf[row * LDX + j]; s += v; s2 += v * v;
        }
        s  += __shfl_xor(s, 1);  s  += __shfl_xor(s, 2);
        s2 += __shfl_xor(s2, 1); s2 += __shfl_xor(s2, 2);
        float m = s * (1.f / CC), var = s2 * (1.f / CC) - m * m;
        float rstd = rsqrtf(var + 1e-5f);
        for (int j = part * 32; j < part * 32 + 32; ++j)
            abuf[row * LDA + j] = (__bf16)((xf[row * LDX + j] - m) * rstd * ln2_g[j] + ln2_b[j]);
    }
    __syncthreads();

    // ---- phase 5: FFN, 4 hidden panels, double-buffered DMA prefetch ------
    __bf16* hbuf = qbuf;                // reuse
    v8f cacc[4]; ZERO4(cacc);
    for (int p = 0; p < 4; ++p) {
        if (p < 3) {                    // prefetch next panel into other pair
            int np = (p + 1) & 1;
            stage_panel(wpan[2 * np],     ws + W1T + (p + 1) * 128 * 128, 128);
            stage_panel(wpan[2 * np + 1], ws + W2T + (p + 1) * 128,       512);
        }
        // hh = relu(h @ w1_panel + b1_panel)
        {
            v8f acc[4]; ZERO4(acc);
            gemm_64x128(abuf, wpan[2 * (p & 1)], acc, wv);
            int n = wv * 16 + (lane & 15);
            float bb1 = b1[p * 128 + n];
#pragma unroll
            for (int tm = 0; tm < 4; ++tm) {
                int m = tm * 16 + ((lane >> 4) << 3);
#pragma unroll
                for (int r = 0; r < 8; ++r)
                    hbuf[(m + r) * LDA + n] = (__bf16)fmaxf(acc[tm][r] + bb1, 0.f);
            }
        }
        __syncthreads();                // hbuf visible
        // out += hh @ w2_panel (register accumulation across panels)
        gemm_64x128(hbuf, wpan[2 * (p & 1) + 1], cacc, wv);
        async_wait_all();               // next panel weights landed
        __syncthreads();
    }
    {   // FFN epilogue: residual + b2
        int n = wv * 16 + (lane & 15);
        float bb2 = b2[n];
#pragma unroll
        for (int tm = 0; tm < 4; ++tm) {
            int m = tm * 16 + ((lane >> 4) << 3);
#pragma unroll
            for (int r = 0; r < 8; ++r) xf[(m + r) * LDX + n] += cacc[tm][r] + bb2;
        }
    }
    __syncthreads();

    // ---- phase 6: MoE bias router (tiny; plain fp32) ----------------------
    if (tid < 128) {                    // column means over T
        float s = 0.f;
        for (int t = 0; t < TT; ++t) s += xf[t * LDX + tid];
        rscr[tid] = s * (1.f / TT);
    }
    __syncthreads();
    if (tid < 64) {                     // hr = relu(r @ rw1 + rb1)
        float s = rb1[tid];
        for (int c = 0; c < 128; ++c) s += rscr[c] * rw1[c * 64 + tid];
        rscr[128 + tid] = fmaxf(s, 0.f);
    }
    __syncthreads();
    if (tid < 8) {                      // logits = hr @ rw2 + rb2
        float s = rb2[tid];
        for (int j = 0; j < 64; ++j) s += rscr[128 + j] * rw2[j * 8 + tid];
        rscr[192 + tid] = s;
    }
    __syncthreads();
    if (tid == 0) {                     // top-2 + softmax (others -> ~0)
        float l[8];
        for (int e = 0; e < 8; ++e) l[e] = rscr[192 + e];
        int i1 = 0;
        for (int e = 1; e < 8; ++e) if (l[e] > l[i1]) i1 = e;
        int i2 = -1;
        for (int e = 0; e < 8; ++e) if (e != i1 && (i2 < 0 || l[e] > l[i2])) i2 = e;
        float m  = fmaxf(l[i1], l[i2]);
        float e1 = __expf(l[i1] - m), e2 = __expf(l[i2] - m);
        float inv = 1.f / (e1 + e2);
        for (int e = 0; e < 8; ++e) rscr[200 + e] = 0.f;
        rscr[200 + i1] = e1 * inv;
        rscr[200 + i2] = e2 * inv;
    }
    __syncthreads();
    if (tid < 128) {                    // bias = scale * (ew @ expert_biases)
        float s = 0.f;
        for (int e = 0; e < 8; ++e) s += rscr[200 + e] * eb[e * 128 + tid];
        rscr[208 + tid] = bscale[0] * s;
    }
    __syncthreads();

    // ---- write output -----------------------------------------------------
    float* ob = out + (size_t)bidx * TT * CC;
    for (int i = tid; i < TT * CC; i += 256)
        ob[i] = xf[(i >> 7) * LDX + (i & 127)] + rscr[208 + (i & 127)];
}

// ---------------------------------------------------------------------------
extern "C" void kernel_launch(void* const* d_in, const int* in_sizes, int n_in,
                              void* d_out, int out_size, void* d_ws, size_t ws_size,
                              hipStream_t stream) {
    (void)in_sizes; (void)n_in; (void)out_size; (void)ws_size;
    const float* x      = (const float*)d_in[0];
    const float* ln1_g  = (const float*)d_in[1];
    const float* ln1_b  = (const float*)d_in[2];
    const float* wq     = (const float*)d_in[3];
    const float* wk     = (const float*)d_in[4];
    const float* wvv    = (const float*)d_in[5];
    const float* wproj  = (const float*)d_in[6];
    const float* b_proj = (const float*)d_in[7];
    const float* ln2_g  = (const float*)d_in[8];
    const float* ln2_b  = (const float*)d_in[9];
    const float* w1     = (const float*)d_in[10];
    const float* b1     = (const float*)d_in[11];
    const float* w2     = (const float*)d_in[12];
    const float* b2     = (const float*)d_in[13];
    const float* rw1    = (const float*)d_in[14];
    const float* rb1    = (const float*)d_in[15];
    const float* rw2    = (const float*)d_in[16];
    const float* rb2    = (const float*)d_in[17];
    const float* eb     = (const float*)d_in[18];
    const float* bscale = (const float*)d_in[19];
    __bf16* ws = (__bf16*)d_ws;

    prep_weights<<<(WS_ELEMS + 255) / 256, 256, 0, stream>>>(wq, wk, wvv, wproj, w1, w2, ws);
    block_kernel<<<4096, 256, 0, stream>>>(x, ln1_g, ln1_b, b_proj, ln2_g, ln2_b,
                                           b1, b2, rw1, rb1, rw2, rb2, eb, bscale,
                                           ws, (float*)d_out);
}
